// GraphInterConnection_13881334301357
// MI455X (gfx1250) — matrix operand order repaired and checked by reading, compiled, not run on hardware
//
#include <hip/hip_runtime.h>
#include <hip/hip_bf16.h>
#include <math.h>

// Problem constants (fixed by the reference).
#define Pn 16
#define Qn 16
#define Bn 4
#define Cn 256
#define Nn 256   // H*H = 16*16

typedef __attribute__((ext_vector_type(16))) __bf16 v16bf;
typedef __attribute__((ext_vector_type(8)))  float  v8f;
typedef __attribute__((ext_vector_type(4)))  int    v4i;

static __device__ __forceinline__ unsigned short f2bfbits(float f) {
  unsigned u = __builtin_bit_cast(unsigned, f);
  u += 0x7FFFu + ((u >> 16) & 1u);            // round-to-nearest-even
  return (unsigned short)(u >> 16);
}
static __device__ __forceinline__ __bf16 f2bf(float f) {
  unsigned short s = f2bfbits(f);
  return __builtin_bit_cast(__bf16, s);
}
static __device__ __forceinline__ float bf2f(__bf16 b) {
  unsigned short s = __builtin_bit_cast(unsigned short, b);
  unsigned u = ((unsigned)s) << 16;
  return __builtin_bit_cast(float, u);
}

// ---- Async global->LDS staging (CDNA5 ASYNC path, with safe fallback) ------
#if defined(__HIP_DEVICE_COMPILE__) && __has_builtin(__builtin_amdgcn_global_load_async_to_lds_b128)
#define HAS_ASYNC_LDS 1
#else
#define HAS_ASYNC_LDS 0
#endif

// Copy 128KB (N*C bf16) from global to LDS, 256 threads, 16B per thread-iter.
static __device__ __forceinline__ void copy_g2lds_128k(char* lds_dst,
                                                       const char* gsrc,
                                                       int tid) {
#if HAS_ASYNC_LDS
  auto gp = (__attribute__((address_space(1))) v4i*)gsrc;
  auto lp = (__attribute__((address_space(3))) v4i*)lds_dst;
#pragma unroll 4
  for (int it = 0; it < 32; ++it) {
    const int idx = it * 256 + tid;   // v4i elements (16 bytes each)
    __builtin_amdgcn_global_load_async_to_lds_b128(gp + idx, lp + idx, 0, 0);
  }
#else
  const uint4* s = (const uint4*)gsrc;
  uint4* d = (uint4*)lds_dst;
#pragma unroll 4
  for (int it = 0; it < 32; ++it) d[it * 256 + tid] = s[it * 256 + tid];
#endif
}

static __device__ __forceinline__ void wait_async_copies() {
#if HAS_ASYNC_LDS
#if __has_builtin(__builtin_amdgcn_s_wait_asynccnt)
  __builtin_amdgcn_s_wait_asynccnt(0);
#else
  asm volatile("s_wait_asynccnt 0x0" ::: "memory");
#endif
#endif
}

// ---- WMMA bf16 fragment loaders (wave32, 16x16x32) -------------------------
// A matrix 16x32 (M x K): interleaved K packing per ISA table:
//   lane = half*16 + m;  VGPR v<4: K = 2v + 8*half ; v>=4: K = 8 + 2v + 8*half
static __device__ __forceinline__ v16bf load_a_frag(const __bf16* __restrict__ base,
                                                    int row0, int k0, int ld) {
  const int lane = threadIdx.x & 31;
  const int half = lane >> 4;
  const __bf16* p = base + (row0 + (lane & 15)) * ld + k0 + half * 8;
  v16bf f;
#pragma unroll
  for (int v = 0; v < 8; ++v) {
    const int kb = (v < 4) ? (2 * v) : (8 + 2 * v);
    f[2 * v]     = p[kb];
    f[2 * v + 1] = p[kb + 1];
  }
  return f;
}

// B matrix 32x16 (K x N): interleaved K packing, same as A (VGPR v<4: K = 2v + 8*half ; v>=4: K = 8 + 2v + 8*half).
// base stores B transposed: base[col][k] row-major with stride ld.
static __device__ __forceinline__ v16bf load_b_frag(const __bf16* __restrict__ base,
                                                    int col0, int k0, int ld) {
  const int lane = threadIdx.x & 31;
  const int half = lane >> 4;
  const __bf16* p = base + (col0 + (lane & 15)) * ld + k0 + half * 8;
  v16bf f;
#pragma unroll
  for (int v = 0; v < 8; ++v) {
    const int kb = (v < 4) ? (2 * v) : (8 + 2 * v);
    f[2 * v]     = p[kb];
    f[2 * v + 1] = p[kb + 1];
  }
  return f;
}

// ---- Prep: bf16 copies of tf in [C,N] and transposed [N,C] layouts ---------
__global__ void __launch_bounds__(256)
prep_convert(const float* __restrict__ temp,
             __bf16* __restrict__ tfB,    // [PB][C][N]
             __bf16* __restrict__ tfTB) { // [PB][N][C]
  const int idx = blockIdx.x * 256 + threadIdx.x;  // over P*B*C*N
  const float v = temp[idx];
  const __bf16 bv = f2bf(v);
  tfB[idx] = bv;
  const int n  = idx & (Nn - 1);
  const int c  = (idx >> 8) & (Cn - 1);
  const int pb = idx >> 16;
  tfTB[((size_t)pb * Nn + n) * Cn + c] = bv;
}

// ---- Prep: corr[pb][i][d] = sum_c We[d][c]*tf[c][i] (bf16) and
//            g[pb][i] = sum_c Wg[c]*tf[c][i] (f32) ---------------------------
__global__ void __launch_bounds__(256)
prep_corr(const float* __restrict__ temp, const float* __restrict__ We,
          const float* __restrict__ Wg, __bf16* __restrict__ corrB,
          float* __restrict__ g) {
  __shared__ float col[Cn];
  __shared__ float red[256];
  const int blk = blockIdx.x;          // pb*N + i
  const int i   = blk & (Nn - 1);
  const int pb  = blk >> 8;
  const int d   = threadIdx.x;
  col[d] = temp[((size_t)(pb * Cn + d)) * Nn + i];
  __syncthreads();
  float s = 0.f;
  const float* wrow = We + d * Cn;
  for (int c = 0; c < Cn; ++c) s += wrow[c] * col[c];
  corrB[(size_t)blk * Cn + d] = f2bf(s);
  red[d] = Wg[d] * col[d];
  __syncthreads();
  for (int st = 128; st > 0; st >>= 1) {
    if (d < st) red[d] += red[d + st];
    __syncthreads();
  }
  if (d == 0) g[blk] = red[0];
}

__global__ void __launch_bounds__(256)
zero_out(float* __restrict__ out) {
  out[blockIdx.x * 256 + threadIdx.x] = 0.f;
}

// ---- Main fused co-attention kernel: one block per (p,q,b) -----------------
// LDS: SL 128KB (S^T bf16) | TS 128KB (staged operand) | mask 1KB  = 257KB
__global__ void __launch_bounds__(256)
coatt_kernel(const __bf16* __restrict__ corrB,  // [PB][N][C]
             const __bf16* __restrict__ tfB,    // [PB][C][N]
             const __bf16* __restrict__ tfTB,   // [PB][N][C]
             const float*  __restrict__ g,      // [PB][N]
             const int*    __restrict__ pnum,
             float* __restrict__ out) {         // [PB][C][N]
  extern __shared__ char smem[];
  __bf16* SL   = (__bf16*)smem;                  // S[j][i] bf16, 128KB
  __bf16* TS   = (__bf16*)(smem + 131072);       // staged tfT (B) / tf (A), 128KB
  float*  mask = (float*)(smem + 262144);        // 256 f32

  const int q = blockIdx.x, p = blockIdx.y, b = blockIdx.z;
  const int pb = p * Bn + b;
  const int qb = q * Bn + b;

  const int tid  = threadIdx.x;
  const int wave = tid >> 5;
  const int lane = tid & 31;
  const int half = lane >> 4;
  const int l15  = lane & 15;

  const __bf16* corrP = corrB + (size_t)pb * Nn * Cn;  // GEMM1 A: [i][d]
  const __bf16* tfTP  = tfTB  + (size_t)qb * Nn * Cn;  // GEMM1 B: [j][d]
  const __bf16* tfP   = tfB   + (size_t)qb * Cn * Nn;  // GEMM2 A: [c][i]
  const float*  gP    = g + qb * Nn;

  // Stage tfT (GEMM1 B operand) into LDS via async copy.
  copy_g2lds_128k((char*)TS, (const char*)tfTP, tid);
  wait_async_copies();
  __syncthreads();

  // Phase A: A[i][j] = sum_d corr[i][d]*tf[d][j]; register softmax over j;
  // SL[j][i] = softmax(A)[i][j] in bf16 (transposed: GEMM2 B operand).
  for (int chunk = 0; chunk < 2; ++chunk) {
    const int i0 = (chunk * 8 + wave) * 16;
    v8f acc[16];
#pragma unroll
    for (int jt = 0; jt < 16; ++jt)
      acc[jt] = (v8f){0.f, 0.f, 0.f, 0.f, 0.f, 0.f, 0.f, 0.f};

    for (int kk = 0; kk < Cn; kk += 32) {
      const v16bf a = load_a_frag(corrP, i0, kk, Cn);
#pragma unroll
      for (int jt = 0; jt < 16; ++jt) {
        const v16bf bb = load_b_frag(TS, jt * 16, kk, Cn);
        acc[jt] = __builtin_amdgcn_wmma_f32_16x16x32_bf16(false, a, false, bb,
                                                          (short)0, acc[jt],
                                                          false, false);
      }
    }

    // Register softmax: VGPR r of each tile holds row (r + 8*half); the 256
    // columns of a row are the 16 tiles x 16 lanes of this half.
    float inv[8];
#pragma unroll
    for (int r = 0; r < 8; ++r) {
      float mx = acc[0][r];
#pragma unroll
      for (int jt = 1; jt < 16; ++jt) mx = fmaxf(mx, acc[jt][r]);
      mx = fmaxf(mx, __shfl_xor(mx, 1, 32));
      mx = fmaxf(mx, __shfl_xor(mx, 2, 32));
      mx = fmaxf(mx, __shfl_xor(mx, 4, 32));
      mx = fmaxf(mx, __shfl_xor(mx, 8, 32));
      float s = 0.f;
#pragma unroll
      for (int jt = 0; jt < 16; ++jt) {
        const float e = __expf(acc[jt][r] - mx);
        acc[jt][r] = e;
        s += e;
      }
      s += __shfl_xor(s, 1, 32);
      s += __shfl_xor(s, 2, 32);
      s += __shfl_xor(s, 4, 32);
      s += __shfl_xor(s, 8, 32);
      inv[r] = 1.f / s;
    }
    // Pack 8 consecutive i per lane per tile -> one ds_store_b128.
#pragma unroll
    for (int jt = 0; jt < 16; ++jt) {
      uint4 w;
      w.x = (unsigned)f2bfbits(acc[jt][0] * inv[0]) |
            ((unsigned)f2bfbits(acc[jt][1] * inv[1]) << 16);
      w.y = (unsigned)f2bfbits(acc[jt][2] * inv[2]) |
            ((unsigned)f2bfbits(acc[jt][3] * inv[3]) << 16);
      w.z = (unsigned)f2bfbits(acc[jt][4] * inv[4]) |
            ((unsigned)f2bfbits(acc[jt][5] * inv[5]) << 16);
      w.w = (unsigned)f2bfbits(acc[jt][6] * inv[6]) |
            ((unsigned)f2bfbits(acc[jt][7] * inv[7]) << 16);
      const int col = jt * 16 + l15;                   // j index
      *(uint4*)((char*)SL + col * (Nn * 2) + (i0 + 8 * half) * 2) = w;
    }
  }
  __syncthreads();   // SL complete; TS reads done

  // Stage tf (GEMM2 A operand) into TS -- async copy overlaps gate compute.
  copy_g2lds_128k((char*)TS, (const char*)tfP, tid);

  // Gate: mask[j] = sigmoid(sum_i g[i] * S[j][i])  (needs only S, not att)
  {
    const int j = tid;
    float m = 0.f;
    for (int i = 0; i < Nn; ++i) m += gP[i] * bf2f(SL[j * Nn + i]);
    mask[j] = 1.f / (1.f + __expf(-m));
  }
  wait_async_copies();
  __syncthreads();

  const float scale = 1.0f / (float)(*pnum);

  // Phase B: att[c][j] = sum_i tf[c][i]*S[j][i]; gate; atomic accumulate.
  for (int iter = 0; iter < 32; ++iter) {
    const int t  = wave + (iter << 3);  // 256 tiles across 8 waves
    const int c0 = (t >> 4) * 16;
    const int j0 = (t & 15) * 16;
    v8f acc = {0.f, 0.f, 0.f, 0.f, 0.f, 0.f, 0.f, 0.f};
#pragma unroll
    for (int kk = 0; kk < Nn; kk += 32) {
      const v16bf a  = load_a_frag(TS, c0, kk, Nn);
      const v16bf bb = load_b_frag(SL, j0, kk, Nn);
      acc = __builtin_amdgcn_wmma_f32_16x16x32_bf16(false, a, false, bb,
                                                    (short)0, acc, false, false);
    }
    const float mj = mask[j0 + l15] * scale;
#pragma unroll
    for (int rr = 0; rr < 8; ++rr) {
      const int c = c0 + rr + 8 * half;
      atomicAdd(&out[((size_t)pb * Cn + c) * Nn + j0 + l15], acc[rr] * mj);
    }
  }
}

extern "C" void kernel_launch(void* const* d_in, const int* in_sizes, int n_in,
                              void* d_out, int out_size, void* d_ws, size_t ws_size,
                              hipStream_t stream) {
  (void)in_sizes; (void)n_in; (void)out_size; (void)ws_size;
  const float* temp = (const float*)d_in[0];  // [P,B,C,H,H] == [PB][C][N]
  const float* We   = (const float*)d_in[1];  // [C,C]
  const float* Wg   = (const float*)d_in[2];  // [C]
  const int*   pn   = (const int*)d_in[3];    // patch_num (scalar)
  float* out = (float*)d_out;                 // [P,B,C,H,H]

  // Workspace layout (bytes): corr bf16 | tf bf16 [C,N] | tf bf16 [N,C] | g f32
  char* ws = (char*)d_ws;
  __bf16* corrB = (__bf16*)(ws);               // PB*N*C*2  = 8,388,608
  __bf16* tfB   = (__bf16*)(ws + 8388608);     // PB*C*N*2  = 8,388,608
  __bf16* tfTB  = (__bf16*)(ws + 16777216);    // PB*N*C*2  = 8,388,608
  float*  g     = (float*)(ws + 25165824);     // PB*N*4    = 65,536

  const int elems = Pn * Bn * Cn * Nn;         // 4,194,304
  prep_convert<<<elems / 256, 256, 0, stream>>>(temp, tfB, tfTB);
  prep_corr<<<Pn * Bn * Nn, 256, 0, stream>>>(temp, We, Wg, corrB, g);
  zero_out<<<elems / 256, 256, 0, stream>>>(out);

  dim3 grid(Qn, Pn, Bn);
  const size_t smem = 131072    // SL (bf16 S^T)
                    + 131072    // TS operand staging
                    + 256 * 4;  // mask
  coatt_kernel<<<grid, 256, smem, stream>>>(corrB, tfB, tfTB, g, pn, out);
}
